// SEDenseNet_85495618994705
// MI455X (gfx1250) — matrix operand order repaired and checked by reading, not compiled, not run on hardware
//
#include <hip/hip_runtime.h>

typedef __attribute__((ext_vector_type(16))) __bf16 v16bf;
typedef __attribute__((ext_vector_type(8)))  __bf16 v8bf;
typedef __attribute__((ext_vector_type(8)))  float  v8f;

__device__ __forceinline__ unsigned short f2bf(float f) {
    union { float f; unsigned u; } v; v.f = f;
    unsigned r = (v.u + 0x7FFFu + ((v.u >> 16) & 1u)) >> 16;
    return (unsigned short)r;
}

// Workgroup barrier that fences ONLY the DS counter: LDS stores are visible,
// LDS fragment loads are retired, but global loads stay in flight across it.
__device__ __forceinline__ void wg_barrier_ds_only() {
    asm volatile(
        "s_wait_dscnt 0x0\n\t"
        "s_barrier_signal -1\n\t"
        "s_barrier_wait -1"
        ::: "memory");
}

// ---------------------------------------------------------------------------
// Prep: W (256x1024 f32) -> bf16 (same [o][c] layout), BN -> scale/bias
// ---------------------------------------------------------------------------
__global__ void prep_kernel(const float* __restrict__ w,
                            const float* __restrict__ gamma,
                            const float* __restrict__ beta,
                            const float* __restrict__ mean,
                            const float* __restrict__ var,
                            unsigned short* __restrict__ wb,
                            float* __restrict__ scale,
                            float* __restrict__ bias) {
    if (blockIdx.x < 1024) {
        int i = blockIdx.x * 256 + threadIdx.x;            // 0..262143
        wb[i] = f2bf(w[i]);
    } else {
        int c = (blockIdx.x - 1024) * 256 + threadIdx.x;   // 0..1023
        float s = gamma[c] * rsqrtf(var[c] + 1e-5f);
        scale[c] = s;
        bias[c]  = beta[c] - mean[c] * s;
    }
}

// ---------------------------------------------------------------------------
// Fused: gather/BN/ReLU -> bf16 LDS (double-buffered, software-pipelined)
//        -> WMMA GEMM (K=1024, K-step 64) -> fused 2x2 avg pool
// grid = 32 images * 49 tiles; block = 256 threads (8 wave32)
// ---------------------------------------------------------------------------
__global__ void __launch_bounds__(256)
fused_bn_conv_pool(const float* __restrict__ x,
                   const unsigned short* __restrict__ Wb,
                   const float* __restrict__ scaleG,
                   const float* __restrict__ biasG,
                   float* __restrict__ yOut) {
    // A tile: 64 rows x 64 k (bf16), pitch 72 elems (144B = 9x16B, conflict-free)
    __shared__ __align__(16) unsigned short Atile[2][64 * 72];
    __shared__ float Cstage[64 * 260];
    __shared__ __align__(16) float sSB[2048];   // interleaved {scale, bias}

    const int tid  = threadIdx.x;
    const int wave = tid >> 5;
    const int lane = tid & 31;
    const int half = lane >> 4;
    const int lcol = lane & 15;

    const int bIdx = blockIdx.x;
    const int bImg = bIdx / 49;
    const int tile = bIdx % 49;
    const int ty = tile / 7, tx = tile % 7;
    const int y0 = ty * 8, x0 = tx * 8;

    // --- preload interleaved BN scale/bias into LDS ---
    #pragma unroll
    for (int i = 0; i < 4; ++i) {
        int c = tid + 256 * i;
        sSB[2 * c]     = scaleG[c];
        sSB[2 * c + 1] = biasG[c];
    }

    // --- per-thread loader coordinates (64 rows x 4 k-groups of 16) ---
    const int r  = tid & 63;
    const int kq = tid >> 6;
    const int ly = r >> 3, lx = r & 7;
    const int yy = y0 + ly, xx = x0 + lx;
    int ymod = yy % 7, xmod = xx % 7;
    const int ym = 4 + (yy / 7) * 6 + (ymod < 5 ? ymod : 5);
    const int xm = 4 + (xx / 7) * 6 + (xmod < 5 ? xmod : 5);
    const int offDir = yy * 56 + xx;
    const int offUp  = ym * 56 + xm;
    const size_t xbase = (size_t)bImg * 512 * 3136;

    // --- WMMA tile coordinates ---
    const int mSub   = wave >> 1;
    const int nHalf  = wave & 1;
    const int nBase0 = nHalf * 128;
    const unsigned short* arow0 = &Atile[0][(mSub * 16 + lcol) * 72];
    const unsigned short* arow1 = &Atile[1][(mSub * 16 + lcol) * 72];

    auto load16 = [&](int ks, float* v) {
        const int cbase = ks * 64 + kq * 16;
        const bool up   = (ks >= 8);
        const int csrc  = cbase - (up ? 512 : 0);
        const float* src = x + xbase + (size_t)csrc * 3136 + (up ? offUp : offDir);
        #pragma unroll
        for (int j = 0; j < 16; ++j) v[j] = src[(size_t)j * 3136];
    };

    v8f acc[8];
    #pragma unroll
    for (int nt = 0; nt < 8; ++nt) acc[nt] = (v8f){0.f,0.f,0.f,0.f,0.f,0.f,0.f,0.f};

    float vin[16];
    load16(0, vin);          // prologue prefetch
    __syncthreads();         // sSB visible (full fence ok, once)

    for (int ks = 0; ks < 16; ++ks) {
        const int p = ks & 1;
        const int cbase = ks * 64 + kq * 16;

        // ---- convert prefetched values: BN + ReLU -> packed bf16 ----
        // (this is the only point that waits on the x loads issued at the end
        //  of the previous K-step)
        unsigned tw[8];
        #pragma unroll
        for (int jj = 0; jj < 8; ++jj) {
            float4 q = *(const float4*)&sSB[2 * (cbase + 2 * jj)]; // {s0,b0,s1,b1}
            float a0 = fmaxf(vin[2 * jj]     * q.x + q.y, 0.f);
            float a1 = fmaxf(vin[2 * jj + 1] * q.z + q.w, 0.f);
            tw[jj] = (unsigned)f2bf(a0) | ((unsigned)f2bf(a1) << 16);
        }
        uint4* dst = (uint4*)&Atile[p][r * 72 + kq * 16];
        uint4 u0; u0.x = tw[0]; u0.y = tw[1]; u0.z = tw[2]; u0.w = tw[3];
        uint4 u1; u1.x = tw[4]; u1.y = tw[5]; u1.z = tw[6]; u1.w = tw[7];
        dst[0] = u0;
        dst[1] = u1;

        // DS-only fence + barrier: global loads stay in flight
        wg_barrier_ds_only();

        // ---- two A fragments (k 0..31 and 32..63 of this step) ----
        const unsigned short* arow = p ? arow1 : arow0;
        v8bf l0 = *(const v8bf*)(arow + half * 16);
        v8bf h0 = *(const v8bf*)(arow + half * 16 + 8);
        v8bf l1 = *(const v8bf*)(arow + 32 + half * 16);
        v8bf h1 = *(const v8bf*)(arow + 32 + half * 16 + 8);
        v16bf af0 = __builtin_shufflevector(l0, h0, 0,1,2,3,4,5,6,7,8,9,10,11,12,13,14,15);
        v16bf af1 = __builtin_shufflevector(l1, h1, 0,1,2,3,4,5,6,7,8,9,10,11,12,13,14,15);

        // ---- 16 WMMAs; B-loads are the only loads their waits can touch ----
        #pragma unroll
        for (int nt = 0; nt < 8; ++nt) {
            const unsigned short* bp = Wb
                + (size_t)(nBase0 + nt * 16 + lcol) * 1024
                + ks * 64 + half * 16;
            v16bf b0 = *(const v16bf*)bp;
            v16bf b1 = *(const v16bf*)(bp + 32);
            acc[nt] = __builtin_amdgcn_wmma_f32_16x16x32_bf16(
                false, af0, false, b0, (short)0, acc[nt], false, false);
            acc[nt] = __builtin_amdgcn_wmma_f32_16x16x32_bf16(
                false, af1, false, b1, (short)0, acc[nt], false, false);
        }

        // ---- issue next K-step's x loads LAST: the convert at the top of the next
        //      K-step waits on them, before its barrier ----
        if (ks < 15) load16(ks + 1, vin);
    }

    // ---- stage C (64 x 256 f32) into LDS ----
    #pragma unroll
    for (int nt = 0; nt < 8; ++nt) {
        int n = nBase0 + nt * 16 + lcol;
        #pragma unroll
        for (int j = 0; j < 8; ++j) {
            int m = mSub * 16 + j + 8 * half;
            Cstage[m * 260 + n] = acc[nt][j];
        }
    }
    __syncthreads();

    // ---- fused 2x2 average pool, coalesced float4 stores ----
    #pragma unroll
    for (int it = 0; it < 4; ++it) {
        int task = tid + 256 * it;       // 0..1023
        int n  = task >> 2;
        int py = task & 3;
        float4 o;
        float* op = (float*)&o;
        #pragma unroll
        for (int px = 0; px < 4; ++px) {
            int m00 = py * 16 + px * 2;
            float v = Cstage[m00 * 260 + n] + Cstage[(m00 + 1) * 260 + n]
                    + Cstage[(m00 + 8) * 260 + n] + Cstage[(m00 + 9) * 260 + n];
            op[px] = v * 0.25f;
        }
        size_t oidx = (((size_t)bImg * 256 + n) * 28 + (ty * 4 + py)) * 28 + tx * 4;
        *(float4*)(yOut + oidx) = o;
    }
}

extern "C" void kernel_launch(void* const* d_in, const int* in_sizes, int n_in,
                              void* d_out, int out_size, void* d_ws, size_t ws_size,
                              hipStream_t stream) {
    const float* x     = (const float*)d_in[0];
    const float* gamma = (const float*)d_in[1];
    const float* beta  = (const float*)d_in[2];
    const float* mean  = (const float*)d_in[3];
    const float* var   = (const float*)d_in[4];
    const float* w     = (const float*)d_in[5];

    unsigned short* Wb = (unsigned short*)d_ws;            // 524288 B
    float* scale = (float*)((char*)d_ws + 524288);
    float* bias  = scale + 1024;

    prep_kernel<<<1028, 256, 0, stream>>>(w, gamma, beta, mean, var, Wb, scale, bias);
    fused_bn_conv_pool<<<32 * 49, 256, 0, stream>>>(x, Wb, scale, bias, (float*)d_out);
}
